// BLEEP_17136919511520
// MI455X (gfx1250) — matrix operand order repaired and checked by reading, not compiled, not run on hardware
//
#include <hip/hip_runtime.h>
#include <hip/hip_bf16.h>

#define NQ 2048
#define NREF 100000
#define GENES 3467
#define DIM 512
#define TOPK 50
#define LN_EPS 1e-5f

typedef __attribute__((ext_vector_type(16))) __bf16 v16bf;
typedef __attribute__((ext_vector_type(8)))  float  v8f;
typedef __attribute__((ext_vector_type(4)))  unsigned int v4u;
typedef __attribute__((ext_vector_type(4)))  unsigned int u32x4;
typedef __attribute__((ext_vector_type(8)))  int i32x8;
typedef __attribute__((ext_vector_type(4)))  int i32x4;

union FragBf { v16bf v; v4u q[2]; };

__device__ __forceinline__ unsigned short f2bf(float x) {
  union { float f; unsigned u; } v; v.f = x;
  unsigned r = v.u + 0x7FFFu + ((v.u >> 16) & 1u);
  return (unsigned short)(r >> 16);
}

#if __has_builtin(__builtin_amdgcn_tensor_load_to_lds)
#define USE_TDM 1
#else
#define USE_TDM 0
#endif

// ---- TDM 2D tile load: tile_y(128) x tile_x(32) bf16 elements, LDS row stride 48 ushorts
//      (pad_interval = 16 DWORDs of data, pad_amount = 8 DWORDs -> 64B data + 32B pad per row).
//      rem_x/rem_y clamp against the tensor edge: OOB reads return zero per ISA.
__device__ __forceinline__ void tdm_load_tile(const unsigned short* gsrc, unsigned lds_off,
                                              int stride_elems, int rem_x, int rem_y) {
#if USE_TDM
  unsigned long long ga = (unsigned long long)(size_t)gsrc;
  unsigned tdx = (unsigned)(rem_x < 0 ? 0 : rem_x);
  unsigned tdy = (unsigned)(rem_y < 0 ? 0 : rem_y);
  u32x4 g0;
  g0[0] = 1u;                                            // count=1, is_restore=0, gather off
  g0[1] = lds_off;                                       // lds_addr
  g0[2] = (unsigned)(ga & 0xFFFFFFFFull);                // global_addr[31:0]
  g0[3] = (unsigned)((ga >> 32) & 0x01FFFFFFull) | (2u << 30);  // addr[56:32] | type=2
  i32x8 g1;
  g1[0] = (int)((1u << 16) |                             // data_size = 2B
                (1u << 20) |                             // pad_enable
                (3u << 22) |                             // pad_interval: 16 DWORDs
                (7u << 25));                             // pad_amount: 8 DWORDs
  g1[1] = (int)((tdx & 0xFFFFu) << 16);                  // tensor_dim0[15:0] @ bits 48..63
  g1[2] = (int)(((tdx >> 16) & 0xFFFFu) | ((tdy & 0xFFFFu) << 16));  // td0[31:16] | td1[15:0]
  g1[3] = (int)(((tdy >> 16) & 0xFFFFu) | (32u << 16));  // td1[31:16] | tile_dim0=32
  g1[4] = (int)128u;                                     // tile_dim1=128, tile_dim2=0
  g1[5] = (int)(unsigned)stride_elems;                   // tensor_dim0_stride[31:0]
  g1[6] = 0;                                             // stride[47:32]=0, dim1_stride lo=0
  g1[7] = 0;
  i32x4 z4 = {0, 0, 0, 0};
#if __clang_major__ >= 23
  i32x8 z8 = {0, 0, 0, 0, 0, 0, 0, 0};
  __builtin_amdgcn_tensor_load_to_lds(g0, g1, z4, z4, z8, 0);
#else
  __builtin_amdgcn_tensor_load_to_lds(g0, g1, z4, z4, 0);
#endif
#else
  (void)gsrc; (void)lds_off; (void)stride_elems; (void)rem_x; (void)rem_y;
#endif
}

// manual fallback staging (host pass / no TDM builtin): 128x32 tile, LDS stride 48
__device__ __forceinline__ void stage_manual(unsigned short* dst, const unsigned short* src,
                                             int ld, int rem_y, int rem_x, int tid) {
  #pragma unroll
  for (int it = 0; it < 16; ++it) {
    int e = tid + it * 256;
    int r = e >> 5, c = e & 31;
    unsigned short v = 0;
    if (r < rem_y && c < rem_x) v = src[(long long)r * ld + c];
    dst[r * 48 + c] = v;
  }
}

// ---------- elementwise f32 -> bf16 (vectorized x4) ----------
__global__ __launch_bounds__(256) void k_conv(const float* __restrict__ src,
                                              unsigned short* __restrict__ dst,
                                              long long n4) {
  long long i = (long long)blockIdx.x * 256 + threadIdx.x;
  if (i < n4) {
    const float4 f = ((const float4*)src)[i];
    unsigned lo = (unsigned)f2bf(f.x) | ((unsigned)f2bf(f.y) << 16);
    unsigned hi = (unsigned)f2bf(f.z) | ((unsigned)f2bf(f.w) << 16);
    ((uint2*)dst)[i] = make_uint2(lo, hi);
  }
}

// ---------- convert + transpose: f32 src[R][C] -> bf16 dst[C][R] ----------
__global__ __launch_bounds__(256) void k_convt(const float* __restrict__ src,
                                               unsigned short* __restrict__ dst,
                                               int R, int C) {
  long long i = (long long)blockIdx.x * 256 + threadIdx.x;
  long long n = (long long)R * C;
  if (i < n) {
    int r = (int)(i / C), c = (int)(i % C);
    dst[(long long)c * R + r] = f2bf(src[i]);
  }
}

// ---------- bf16 WMMA GEMM, TDM-staged, double-buffered ----------
// C = A(bf16,[m][k]) @ B(bf16,[n][k])^T
// EPI: 0 = +bias, also emit gout = bf16(gelu(C)) ; 1 = +bias + resid ; 2 = plain
template<int EPI>
__global__ __launch_bounds__(256) void k_gemm(
    const unsigned short* __restrict__ A, int lda,
    const unsigned short* __restrict__ B, int ldb,
    float* __restrict__ C, int ldc,
    const float* __restrict__ bias,
    const float* __restrict__ resid,
    unsigned short* __restrict__ gout,
    int M, int N, int K)
{
  constexpr int BM = 128, BN = 128, BK = 32, LS = 48;
  __shared__ __align__(16) unsigned short As[2][BM * LS];
  __shared__ __align__(16) unsigned short Bs[2][BN * LS];

  const int tid  = threadIdx.x;
  const int m0   = blockIdx.y * BM;
  const int n0   = blockIdx.x * BN;
  const int wave = tid >> 5, lane = tid & 31;
  const int lidx = lane & 15, lhalf = lane >> 4;
  const int rowbase = (wave & 1) * 64;
  const int colbase = (wave >> 1) * 32;

  const v8f vzero = {};
  v8f acc[4][2];
  #pragma unroll
  for (int i = 0; i < 4; ++i)
    #pragma unroll
    for (int j = 0; j < 2; ++j) acc[i][j] = vzero;

  const int nkt = (K + BK - 1) / BK;

  auto stage = [&](int kt, int buf) {
#if USE_TDM
    if (wave == 0) {
      tdm_load_tile(A + (long long)m0 * lda + kt * BK, (unsigned)(size_t)&As[buf][0],
                    lda, K - kt * BK, M - m0);
      tdm_load_tile(B + (long long)n0 * ldb + kt * BK, (unsigned)(size_t)&Bs[buf][0],
                    ldb, K - kt * BK, N - n0);
    }
#else
    stage_manual(&As[buf][0], A + (long long)m0 * lda + kt * BK, lda,
                 M - m0 < BM ? M - m0 : BM, K - kt * BK < BK ? K - kt * BK : BK, tid);
    stage_manual(&Bs[buf][0], B + (long long)n0 * ldb + kt * BK, ldb,
                 N - n0 < BN ? N - n0 : BN, K - kt * BK < BK ? K - kt * BK : BK, tid);
#endif
  };

  // prologue: stage tile 0
  stage(0, 0);
#if USE_TDM
  if (wave == 0) __builtin_amdgcn_s_wait_tensorcnt(0);
#endif
  __syncthreads();

  int cur = 0;
  for (int kt = 0; kt < nkt; ++kt) {
    if (kt + 1 < nkt) stage(kt + 1, cur ^ 1);   // async DMA of next tile

    FragBf a[4], b[2];
    #pragma unroll
    for (int i = 0; i < 4; ++i) {
      const unsigned short* ap = &As[cur][(rowbase + 16 * i + lidx) * LS];
      a[i].q[0] = *(const v4u*)(ap + (lhalf ? 8  : 0));
      a[i].q[1] = *(const v4u*)(ap + (lhalf ? 24 : 16));
    }
    #pragma unroll
    for (int j = 0; j < 2; ++j) {
      const unsigned short* bp = &Bs[cur][(colbase + 16 * j + lidx) * LS];
      b[j].q[0] = *(const v4u*)(bp + (lhalf ? 8  : 0));
      b[j].q[1] = *(const v4u*)(bp + (lhalf ? 24 : 16));
    }
    #pragma unroll
    for (int i = 0; i < 4; ++i)
      #pragma unroll
      for (int j = 0; j < 2; ++j)
        acc[i][j] = __builtin_amdgcn_wmma_f32_16x16x32_bf16(
            false, a[i].v, false, b[j].v, (short)0, acc[i][j], false, false);

    __syncthreads();                 // everyone done reading buf `cur`
#if USE_TDM
    if (kt + 1 < nkt) {
      if (wave == 0) __builtin_amdgcn_s_wait_tensorcnt(0);
      __syncthreads();               // publish staged tile to all waves
    }
#endif
    cur ^= 1;
  }

  #pragma unroll
  for (int i = 0; i < 4; ++i) {
    #pragma unroll
    for (int j = 0; j < 2; ++j) {
      #pragma unroll
      for (int r = 0; r < 8; ++r) {
        int gm = m0 + rowbase + 16 * i + (lhalf ? r + 8 : r);
        int gn = n0 + colbase + 16 * j + lidx;
        if (gm < M && gn < N) {
          float v = acc[i][j][r];
          if (EPI == 0) {
            v += bias[gn];
            float g = 0.5f * v * (1.0f + erff(v * 0.70710678118654752f));
            gout[(long long)gm * ldc + gn] = f2bf(g);
          }
          if (EPI == 1) v += bias[gn] + resid[(long long)gm * ldc + gn];
          C[(long long)gm * ldc + gn] = v;
        }
      }
    }
  }
}

// ---------- LayerNorm + L2-normalize: wave per row (DIM=512 -> 16 per lane) ----------
__global__ __launch_bounds__(256) void k_ln_norm(const float* __restrict__ H,
                                                 float* __restrict__ E,
                                                 unsigned short* __restrict__ Sn,
                                                 const float* __restrict__ gamma,
                                                 const float* __restrict__ beta,
                                                 int rows) {
  int wave = threadIdx.x >> 5, lane = threadIdx.x & 31;
  int row = blockIdx.x * 8 + wave;
  if (row >= rows) return;
  const float* h = H + (long long)row * DIM;
  float x[16], s = 0.f, sq = 0.f;
  #pragma unroll
  for (int j = 0; j < 16; ++j) { x[j] = h[lane + 32 * j]; s += x[j]; sq += x[j] * x[j]; }
  #pragma unroll
  for (int o = 16; o > 0; o >>= 1) { s += __shfl_xor(s, o, 32); sq += __shfl_xor(sq, o, 32); }
  float mu  = s / DIM;
  float var = sq / DIM - mu * mu;
  float inv = rsqrtf(var + LN_EPS);
  float y[16], n2 = 0.f;
  #pragma unroll
  for (int j = 0; j < 16; ++j) {
    int c = lane + 32 * j;
    y[j] = (x[j] - mu) * inv * gamma[c] + beta[c];
    n2 += y[j] * y[j];
  }
  #pragma unroll
  for (int o = 16; o > 0; o >>= 1) n2 += __shfl_xor(n2, o, 32);
  float rn = rsqrtf(n2);
  float* e = E + (long long)row * DIM;
  unsigned short* sn = Sn + (long long)row * DIM;
  #pragma unroll
  for (int j = 0; j < 16; ++j) { int c = lane + 32 * j; e[c] = y[j]; sn[c] = f2bf(y[j] * rn); }
}

// ---------- query L2-normalize -> bf16 ----------
__global__ __launch_bounds__(256) void k_qnorm(const float* __restrict__ Q,
                                               unsigned short* __restrict__ Qn,
                                               int rows) {
  int wave = threadIdx.x >> 5, lane = threadIdx.x & 31;
  int row = blockIdx.x * 8 + wave;
  if (row >= rows) return;
  const float* q = Q + (long long)row * DIM;
  float x[16], n2 = 0.f;
  #pragma unroll
  for (int j = 0; j < 16; ++j) { x[j] = q[lane + 32 * j]; n2 += x[j] * x[j]; }
  #pragma unroll
  for (int o = 16; o > 0; o >>= 1) n2 += __shfl_xor(n2, o, 32);
  float rn = rsqrtf(n2);
  unsigned short* qn = Qn + (long long)row * DIM;
  #pragma unroll
  for (int j = 0; j < 16; ++j) qn[lane + 32 * j] = f2bf(x[j] * rn);
}

// ---------- top-50 per query ----------
__global__ __launch_bounds__(64) void k_topk(const float* __restrict__ sim,
                                             int* __restrict__ outIdx) {
  __shared__ float sv[64 * TOPK];
  __shared__ int   si[64 * TOPK];
  __shared__ float tv[32 * TOPK];
  __shared__ int   ti[32 * TOPK];
  const int t = threadIdx.x;
  const int q = blockIdx.x;
  const float* row = sim + (long long)q * NREF;
  const int base = t * TOPK;
  int cnt = 0;
  for (int j = t; j < NREF; j += 64) {
    float v = row[j];
    if (cnt < TOPK) {
      int p = cnt++;
      while (p > 0 && sv[base + p - 1] < v) {
        sv[base + p] = sv[base + p - 1]; si[base + p] = si[base + p - 1]; --p;
      }
      sv[base + p] = v; si[base + p] = j;
    } else if (v > sv[base + TOPK - 1]) {
      int p = TOPK - 1;
      while (p > 0 && sv[base + p - 1] < v) {
        sv[base + p] = sv[base + p - 1]; si[base + p] = si[base + p - 1]; --p;
      }
      sv[base + p] = v; si[base + p] = j;
    }
  }
  for (int p = cnt; p < TOPK; ++p) { sv[base + p] = -3.0e38f; si[base + p] = 0; }
  __syncthreads();
  for (int s = 32; s >= 1; s >>= 1) {
    if (t < s) {
      int ia = t * TOPK, ib = (t + s) * TOPK, io = t * TOPK;
      int pa = 0, pb = 0;
      for (int o = 0; o < TOPK; ++o) {
        float va = (pa < TOPK) ? sv[ia + pa] : -3.0e38f;
        float vb = (pb < TOPK) ? sv[ib + pb] : -3.0e38f;
        if (va >= vb) { tv[io + o] = va; ti[io + o] = si[ia + pa]; ++pa; }
        else          { tv[io + o] = vb; ti[io + o] = si[ib + pb]; ++pb; }
      }
      for (int o = 0; o < TOPK; ++o) { sv[ia + o] = tv[io + o]; si[ia + o] = ti[io + o]; }
    }
    __syncthreads();
  }
  if (t < TOPK) outIdx[q * TOPK + t] = si[t];
}

// ---------- squared distances ----------
__global__ __launch_bounds__(256) void k_dist(const float* __restrict__ E,
                                              const float* __restrict__ Qf,
                                              const int* __restrict__ idx,
                                              float* __restrict__ dbuf) {
  int q = blockIdx.x, wave = threadIdx.x >> 5, lane = threadIdx.x & 31;
  const float* qf = Qf + (long long)q * DIM;
  for (int k = wave; k < TOPK; k += 8) {
    int id = idx[q * TOPK + k];
    const float* e = E + (long long)id * DIM;
    float s = 0.f;
    #pragma unroll
    for (int j = 0; j < 16; ++j) {
      float d = e[lane + 32 * j] - qf[lane + 32 * j];
      s += d * d;
    }
    #pragma unroll
    for (int o = 16; o > 0; o >>= 1) s += __shfl_xor(s, o, 32);
    if (lane == 0) dbuf[q * TOPK + k] = s;
  }
}

// ---------- normalized weights ----------
__global__ __launch_bounds__(32) void k_weights(const float* __restrict__ dbuf,
                                                float* __restrict__ wn) {
  int q = blockIdx.x, lane = threadIdx.x;
  float d0 = dbuf[q * TOPK];
  int k0i = lane, k1i = lane + 32;
  float w0 = 0.f, w1 = 0.f;
  if (k0i < TOPK) w0 = __expf(-(dbuf[q * TOPK + k0i] - d0 + 1.0f));
  if (k1i < TOPK) w1 = __expf(-(dbuf[q * TOPK + k1i] - d0 + 1.0f));
  float s = w0 + w1;
  #pragma unroll
  for (int o = 16; o > 0; o >>= 1) s += __shfl_xor(s, o, 32);
  float inv = 1.0f / s;
  if (k0i < TOPK) wn[q * TOPK + k0i] = w0 * inv;
  if (k1i < TOPK) wn[q * TOPK + k1i] = w1 * inv;
}

// ---------- weighted gather ----------
__global__ __launch_bounds__(256) void k_pred(const float* __restrict__ ref,
                                              const int* __restrict__ idx,
                                              const float* __restrict__ wn,
                                              float* __restrict__ out) {
  __shared__ float lw[TOPK];
  __shared__ int   li[TOPK];
  int q = blockIdx.x;
  if (threadIdx.x < TOPK) {
    lw[threadIdx.x] = wn[q * TOPK + threadIdx.x];
    li[threadIdx.x] = idx[q * TOPK + threadIdx.x];
  }
  __syncthreads();
  for (int g = threadIdx.x; g < GENES; g += 256) {
    float acc = 0.f;
    #pragma unroll
    for (int k = 0; k < TOPK; ++k)
      acc += ref[(long long)li[k] * GENES + g] * lw[k];
    out[(long long)q * GENES + g] = acc;
  }
}

extern "C" void kernel_launch(void* const* d_in, const int* in_sizes, int n_in,
                              void* d_out, int out_size, void* d_ws, size_t ws_size,
                              hipStream_t stream) {
  const float* img   = (const float*)d_in[0];
  const float* sref  = (const float*)d_in[1];
  const float* W1    = (const float*)d_in[2];
  const float* b1    = (const float*)d_in[3];
  const float* W2    = (const float*)d_in[4];
  const float* b2    = (const float*)d_in[5];
  const float* gamma = (const float*)d_in[6];
  const float* beta  = (const float*)d_in[7];
  float* out = (float*)d_out;

  char* w = (char*)d_ws;
  size_t off = 0;
  auto alloc = [&](size_t bytes) { char* p = w + off; off += (bytes + 255) & ~(size_t)255; return p; };
  unsigned short* Xbf = (unsigned short*)alloc((size_t)NREF * GENES * 2); // bf16 spot_ref
  unsigned short* W1T = (unsigned short*)alloc((size_t)DIM * GENES * 2);  // [DIM][GENES]
  unsigned short* W2T = (unsigned short*)alloc((size_t)DIM * DIM * 2);
  float*          P   = (float*)alloc((size_t)NREF * DIM * 4);            // projected
  unsigned short* Gbf = (unsigned short*)alloc((size_t)NREF * DIM * 2);   // bf16 gelu(P)
  float*          E   = (float*)alloc((size_t)NREF * DIM * 4);            // spot_emb
  unsigned short* Sn  = (unsigned short*)alloc((size_t)NREF * DIM * 2);
  unsigned short* Qn  = (unsigned short*)alloc((size_t)NQ * DIM * 2);
  float*          sim = (float*)alloc((size_t)NQ * NREF * 4);
  int*            idx = (int*)alloc((size_t)NQ * TOPK * 4);
  float*          dbf = (float*)alloc((size_t)NQ * TOPK * 4);
  float*          wn  = (float*)alloc((size_t)NQ * TOPK * 4);
  (void)ws_size; (void)in_sizes; (void)n_in; (void)out_size;

  // pre-convert operands to bf16
  {
    long long n4 = (long long)NREF * GENES / 4;
    k_conv<<<(unsigned)((n4 + 255) / 256), 256, 0, stream>>>(sref, Xbf, n4);
  }
  k_convt<<<(GENES * DIM + 255) / 256, 256, 0, stream>>>(W1, W1T, GENES, DIM);
  k_convt<<<(DIM * DIM + 255) / 256, 256, 0, stream>>>(W2, W2T, DIM, DIM);

  // GEMM1: P = X @ W1 + b1 ; Gbf = bf16(gelu(P))
  {
    dim3 g((DIM + 127) / 128, (NREF + 127) / 128);
    k_gemm<0><<<g, 256, 0, stream>>>(Xbf, GENES, W1T, GENES, P, DIM, b1, nullptr, Gbf,
                                     NREF, DIM, GENES);
  }
  // GEMM2: E = gelu(P) @ W2 + b2 + P
  {
    dim3 g((DIM + 127) / 128, (NREF + 127) / 128);
    k_gemm<1><<<g, 256, 0, stream>>>(Gbf, DIM, W2T, DIM, E, DIM, b2, P, nullptr,
                                     NREF, DIM, DIM);
  }
  // LayerNorm + normalize
  k_ln_norm<<<(NREF + 7) / 8, 256, 0, stream>>>(E, E, Sn, gamma, beta, NREF);
  k_qnorm<<<(NQ + 7) / 8, 256, 0, stream>>>(img, Qn, NQ);
  // GEMM3: sim = Qn @ Sn^T
  {
    dim3 g((NREF + 127) / 128, (NQ + 127) / 128);
    k_gemm<2><<<g, 256, 0, stream>>>(Qn, DIM, Sn, DIM, sim, NREF, nullptr, nullptr, nullptr,
                                     NQ, NREF, DIM);
  }
  // top-k, distances, weights, weighted gather
  k_topk<<<NQ, 64, 0, stream>>>(sim, idx);
  k_dist<<<NQ, 256, 0, stream>>>(E, img, idx, dbf);
  k_weights<<<NQ, 32, 0, stream>>>(dbf, wn);
  k_pred<<<NQ, 256, 0, stream>>>(sref, idx, wn, out);
}
